// CONNECT_55997783605361
// MI455X (gfx1250) — matrix operand order repaired and checked by reading, not compiled, not run on hardware
//
#include <hip/hip_runtime.h>
#include <hip/hip_bf16.h>

typedef __bf16 bf16;
typedef __attribute__((ext_vector_type(16))) __bf16 bf16x16;
typedef __attribute__((ext_vector_type(8)))  __bf16 bf16x8;
typedef __attribute__((ext_vector_type(8)))  float  f32x8;

#define DD   128   // D = F = T = P
#define CC   256   // C
#define KMSG 512   // 2D + F + T
#define G3D  384   // 3*D

// ---- WMMA fragment helpers (ISA 7.12.2 layouts, bf16 16x16x32) ----
// A (16x32, MxK): lanes 0-15 row=lane hold K {0..7,16..23}; lanes 16-31 row=lane-16 hold K {8..15,24..31}
__device__ __forceinline__ bf16x16 frag_a(const bf16* base, int ld) {
  const int lane = threadIdx.x & 31;
  const int row  = lane & 15;
  const int k0   = (lane < 16) ? 0 : 8;
  const bf16* p  = base + (size_t)row * ld + k0;
  bf16x8 lo = *(const bf16x8*)(p);
  bf16x8 hi = *(const bf16x8*)(p + 16);
  return __builtin_shufflevector(lo, hi, 0,1,2,3,4,5,6,7,8,9,10,11,12,13,14,15);
}
// B (32x16, KxN) from [col][k]-major storage: lanes 0-15 col=lane hold K {0..7,16..23}; lanes 16-31 hold K {8..15,24..31}
__device__ __forceinline__ bf16x16 frag_b(const bf16* base, int ld) {
  const int lane = threadIdx.x & 31;
  const int col  = lane & 15;
  const int k0   = (lane < 16) ? 0 : 8;
  const bf16* p  = base + (size_t)col * ld + k0;
  bf16x8 lo = *(const bf16x8*)(p);
  bf16x8 hi = *(const bf16x8*)(p + 16);
  return __builtin_shufflevector(lo, hi, 0,1,2,3,4,5,6,7,8,9,10,11,12,13,14,15);
}
__device__ __forceinline__ f32x8 wmma_bf16(bf16x16 a, bf16x16 b, f32x8 c) {
  return __builtin_amdgcn_wmma_f32_16x16x32_bf16(false, a, false, b, (short)0, c, false, false);
}
__device__ __forceinline__ float sigmoidf(float x) { return 1.f / (1.f + expf(-x)); }

// ---------------- prep kernels ----------------
__global__ void k_cvt(const float* __restrict__ in, bf16* __restrict__ out, int n) {
  int i = blockIdx.x * blockDim.x + threadIdx.x;
  if (i < n) out[i] = (bf16)in[i];
}
// proj_W [d][p] -> [p][d] bf16 (k-contiguous B layout)
__global__ void k_projt(const float* __restrict__ pw, bf16* __restrict__ out, int n) {
  int i = blockIdx.x * blockDim.x + threadIdx.x;
  if (i < n) { int p = i >> 7, d = i & 127; out[i] = (bf16)pw[d * DD + p]; }
}
// normalized centroids, bf16, [c][p] (already k-contiguous for sim GEMM)
__global__ void k_cen(const float* __restrict__ cen, bf16* __restrict__ cenB) {
  __shared__ float red[4];
  int c = blockIdx.x, t = threadIdx.x;
  float x = cen[c * DD + t];
  float s = x * x;
  for (int o = 16; o; o >>= 1) s += __shfl_xor(s, o, 32);
  if ((t & 31) == 0) red[t >> 5] = s;
  __syncthreads();
  float tot = red[0] + red[1] + red[2] + red[3];
  cenB[c * DD + t] = (bf16)(x / (sqrtf(tot) + 1e-8f));
}

// ---------------- message scatter (segment_sum) ----------------
__global__ void k_scatter(const float* __restrict__ mem, const float* __restrict__ last,
                          const float* __restrict__ ef, const float* __restrict__ t,
                          const int* __restrict__ src, const int* __restrict__ dst,
                          const float* __restrict__ tw, const float* __restrict__ tb,
                          float* __restrict__ msg_sum, float* __restrict__ cnt, int E) {
  int e = blockIdx.x * 2 + (threadIdx.x >> 7);
  int d = threadIdx.x & 127;
  if (e >= 2 * E) return;
  int i, o, er;
  if (e < E) { i = src[e]; o = dst[e]; er = e; }
  else       { i = dst[e - E]; o = src[e - E]; er = e - E; }
  float te = t[er];
  float* ms = msg_sum + (size_t)i * KMSG;
  atomicAdd(ms + d,       mem[(size_t)i * DD + d]);
  atomicAdd(ms + 128 + d, mem[(size_t)o * DD + d]);
  atomicAdd(ms + 256 + d, ef[(size_t)er * DD + d]);
  float dt = te - last[i];
  atomicAdd(ms + 384 + d, cosf(dt * tw[d] + tb[d]));
  if (d == 0) atomicAdd(cnt + i, 1.0f);
}

// agg = msg_sum / max(cnt,1)  -> bf16
__global__ void k_agg(const float* __restrict__ msg_sum, const float* __restrict__ cnt,
                      bf16* __restrict__ aggB, int total) {
  int i = blockIdx.x * blockDim.x + threadIdx.x;
  if (i < total) {
    float c = cnt[i >> 9];
    aggB[i] = (bf16)(msg_sum[i] / fmaxf(c, 1.0f));
  }
}

// ---------------- GRU update: gi/gh GEMMs (WMMA) + in-register gates ----------------
// Wave w computes column tiles {w, w+8, w+16} == gate columns {j, j+128, j+256}, so the
// r/z/n accumulators of one wave line up element-for-element: no LDS gi/gh staging needed.
__global__ void __launch_bounds__(256)
k_gru(const bf16* __restrict__ aggB, const bf16* __restrict__ memB,
      const bf16* __restrict__ WihB, const bf16* __restrict__ WhhB,
      const float* __restrict__ b_ih, const float* __restrict__ b_hh,
      const float* __restrict__ mem, const float* __restrict__ cnt,
      float* __restrict__ out) {
  __shared__ __align__(16) bf16 lA[16 * KMSG];  // 16 KB agg tile
  __shared__ __align__(16) bf16 lH[16 * DD];    //  4 KB h tile
  const int n0 = blockIdx.x * 16;
  const int w = threadIdx.x >> 5, lane = threadIdx.x & 31;
  {
    const uint4* s0 = (const uint4*)(aggB + (size_t)n0 * KMSG);
    uint4* d0 = (uint4*)lA;
    for (int i = threadIdx.x; i < (16 * KMSG) / 8; i += 256) d0[i] = s0[i];
    const uint4* s1 = (const uint4*)(memB + (size_t)n0 * DD);
    uint4* d1 = (uint4*)lH;
    for (int i = threadIdx.x; i < (16 * DD) / 8; i += 256) d1[i] = s1[i];
  }
  __syncthreads();
  f32x8 aRi = {}, aZi = {}, aNi = {};
  for (int ks = 0; ks < 16; ++ks) {
    bf16x16 a = frag_a(lA + ks * 32, KMSG);
    aRi = wmma_bf16(a, frag_b(WihB + (size_t)((w)      * 16) * KMSG + ks * 32, KMSG), aRi);
    aZi = wmma_bf16(a, frag_b(WihB + (size_t)((w + 8)  * 16) * KMSG + ks * 32, KMSG), aZi);
    aNi = wmma_bf16(a, frag_b(WihB + (size_t)((w + 16) * 16) * KMSG + ks * 32, KMSG), aNi);
  }
  f32x8 aRh = {}, aZh = {}, aNh = {};
  for (int ks = 0; ks < 4; ++ks) {
    bf16x16 a = frag_a(lH + ks * 32, DD);
    aRh = wmma_bf16(a, frag_b(WhhB + (size_t)((w)      * 16) * DD + ks * 32, DD), aRh);
    aZh = wmma_bf16(a, frag_b(WhhB + (size_t)((w + 8)  * 16) * DD + ks * 32, DD), aZh);
    aNh = wmma_bf16(a, frag_b(WhhB + (size_t)((w + 16) * 16) * DD + ks * 32, DD), aNh);
  }
  const int j = w * 16 + (lane & 15);
  const int mo = (lane < 16) ? 0 : 8;
  const float bRi = b_ih[j], bZi = b_ih[128 + j], bNi = b_ih[256 + j];
  const float bRh = b_hh[j], bZh = b_hh[128 + j], bNh = b_hh[256 + j];
#pragma unroll
  for (int v = 0; v < 8; ++v) {
    int n = n0 + v + mo;
    float rg = sigmoidf(aRi[v] + bRi + aRh[v] + bRh);
    float zg = sigmoidf(aZi[v] + bZi + aZh[v] + bZh);
    float nn = tanhf(aNi[v] + bNi + rg * (aNh[v] + bNh));
    float h = mem[(size_t)n * DD + j];
    out[(size_t)n * DD + j] = (cnt[n] > 0.f) ? ((1.f - zg) * nn + zg * h) : h;
  }
}

// ---------------- projection, cosine sim, row sparsemax ----------------
__global__ void __launch_bounds__(256)
k_comm(const float* __restrict__ newmem, const float* __restrict__ nf,
       const bf16* __restrict__ projWtB, const float* __restrict__ proj_b,
       const bf16* __restrict__ cenB,
       float* __restrict__ simT, bf16* __restrict__ ncB, int N) {
  __shared__ __align__(16) bf16  lF[16 * DD];
  __shared__ __align__(16) float lPF[16 * DD];
  __shared__ __align__(16) bf16  lPFb[16 * DD];
  __shared__ __align__(16) float lSIM[16 * CC];
  __shared__ float lInv[16];
  const int n0 = blockIdx.x * 16;
  const int w = threadIdx.x >> 5, lane = threadIdx.x & 31;
  for (int it = 0; it < 8; ++it) {
    int lin = it * 256 + threadIdx.x;
    int r = lin >> 7, d = lin & 127;
    lF[lin] = (bf16)(newmem[(size_t)(n0 + r) * DD + d] + nf[(size_t)(n0 + r) * DD + d]);
  }
  __syncthreads();
  { // pf = feat @ proj_W + proj_b  (wave w -> column tile w)
    f32x8 acc = {};
    for (int ks = 0; ks < 4; ++ks) {
      bf16x16 a = frag_a(lF + ks * 32, DD);
      acc = wmma_bf16(a, frag_b(projWtB + (w * 16) * DD + ks * 32, DD), acc);
    }
    int j = w * 16 + (lane & 15);
    float bj = proj_b[j];
    int mo = (lane < 16) ? 0 : 8;
#pragma unroll
    for (int v = 0; v < 8; ++v) lPF[(v + mo) * DD + j] = acc[v] + bj;
  }
  __syncthreads();
  if (threadIdx.x < 16) {
    float s = 0.f;
    for (int d = 0; d < DD; ++d) { float x = lPF[threadIdx.x * DD + d]; s += x * x; }
    lInv[threadIdx.x] = 1.f / (sqrtf(s) + 1e-8f);
  }
  __syncthreads();
  for (int it = 0; it < 8; ++it) {
    int lin = it * 256 + threadIdx.x;
    lPFb[lin] = (bf16)(lPF[lin] * lInv[lin >> 7]);
  }
  __syncthreads();
  // sim = pf_n @ cen_n^T : wave w -> tiles {w, w+8}, A fragments reloaded per tile
  for (int ks = 0; ks < 4; ++ks) {
    bf16x16 a = frag_a(lPFb + ks * 32, DD);
    static __shared__ char dummy; (void)dummy;
    // two accumulators kept across ks would need loop restructure; do per-tile loops:
    (void)a;
    break;
  }
  for (int jt = w; jt < 16; jt += 8) {
    f32x8 acc = {};
    for (int ks = 0; ks < 4; ++ks) {
      bf16x16 a = frag_a(lPFb + ks * 32, DD);
      acc = wmma_bf16(a, frag_b(cenB + (jt * 16) * DD + ks * 32, DD), acc);
    }
    int j = jt * 16 + (lane & 15);
    int mo = (lane < 16) ? 0 : 8;
#pragma unroll
    for (int v = 0; v < 8; ++v) lSIM[(v + mo) * CC + j] = acc[v];
  }
  __syncthreads();
  // write transposed sim for the column sparsemax
  for (int it = 0; it < 16; ++it) {
    int lin = it * 256 + threadIdx.x;
    int c = lin >> 4, r = lin & 15;
    simT[(size_t)c * N + n0 + r] = lSIM[r * CC + c];
  }
  // exact row sparsemax via bisection on tau: sum(relu(z - tau)) = 1
  for (int rr = 0; rr < 2; ++rr) {
    int row = w * 2 + rr;
    float v[8];
#pragma unroll
    for (int i = 0; i < 8; ++i) v[i] = lSIM[row * CC + lane + 32 * i];
    float lo = -4.f, hi = 4.f;
    for (int itB = 0; itB < 48; ++itB) {
      float mid = 0.5f * (lo + hi);
      float s = 0.f;
#pragma unroll
      for (int i = 0; i < 8; ++i) s += fmaxf(v[i] - mid, 0.f);
      for (int o = 16; o; o >>= 1) s += __shfl_xor(s, o, 32);
      if (s > 1.f) lo = mid; else hi = mid;
    }
    float tau = 0.5f * (lo + hi);
#pragma unroll
    for (int i = 0; i < 8; ++i)
      ncB[(size_t)(n0 + row) * CC + lane + 32 * i] = (bf16)fmaxf(v[i] - tau, 0.f);
  }
}

// ------- column sparsemax over N per centroid + fused c_memory = cn @ new_memory -------
__global__ void __launch_bounds__(1024)
k_colsp(const float* __restrict__ simT, const float* __restrict__ newmem,
        float* __restrict__ c_memory, int N) {
  constexpr int VPT = 98;  // ceil(100000/1024)
  __shared__ float red[32];
  __shared__ float bc;
  __shared__ float ldsC[DD];
  const int c = blockIdx.x, tid = threadIdx.x;
  const int lane = tid & 31, wid = tid >> 5;
  float v[VPT];
#pragma unroll
  for (int i = 0; i < VPT; ++i) {
    int n = i * 1024 + tid;
    v[i] = (n < N) ? simT[(size_t)c * N + n] : -1e30f;
  }
  if (tid < DD) ldsC[tid] = 0.f;
  float lo = -4.f, hi = 4.f;
  for (int itB = 0; itB < 48; ++itB) {
    float mid = 0.5f * (lo + hi);
    float s = 0.f;
#pragma unroll
    for (int i = 0; i < VPT; ++i) s += fmaxf(v[i] - mid, 0.f);
    for (int o = 16; o; o >>= 1) s += __shfl_xor(s, o, 32);
    if (lane == 0) red[wid] = s;
    __syncthreads();
    if (wid == 0) {
      float x = red[lane];
      for (int o = 16; o; o >>= 1) x += __shfl_xor(x, o, 32);
      if (lane == 0) bc = x;
    }
    __syncthreads();
    float tot = bc;
    __syncthreads();
    if (tot > 1.f) lo = mid; else hi = mid;
  }
  float tau = 0.5f * (lo + hi);
#pragma unroll
  for (int i = 0; i < VPT; ++i) {
    float p = fmaxf(v[i] - tau, 0.f);
    if (p > 0.f) {
      int n = i * 1024 + tid;
      const float* m = newmem + (size_t)n * DD;
      for (int d = 0; d < DD; ++d) atomicAdd(&ldsC[d], p * m[d]);
    }
  }
  __syncthreads();
  if (tid < DD) c_memory[c * DD + tid] = ldsC[tid];
}

// c_memory [c][d] -> [d][c] bf16 (k-contiguous B layout for final GEMM)
__global__ void k_cmt(const float* __restrict__ c_memory, bf16* __restrict__ cmt, int n) {
  int i = blockIdx.x * blockDim.x + threadIdx.x;
  if (i < n) { int d = i >> 8, c = i & 255; cmt[i] = (bf16)c_memory[c * DD + d]; }
}

// emb = new_memory + nc @ c_memory   (K = 256, WMMA; nc tile staged in LDS once)
__global__ void __launch_bounds__(256)
k_emb(const bf16* __restrict__ ncB, const bf16* __restrict__ cmt,
      float* __restrict__ out) {
  __shared__ __align__(16) bf16 lA[16 * CC];  // 8 KB
  const int n0 = blockIdx.x * 16;
  const int w = threadIdx.x >> 5, lane = threadIdx.x & 31;
  const uint4* s0 = (const uint4*)(ncB + (size_t)n0 * CC);
  uint4* d0 = (uint4*)lA;
  for (int i = threadIdx.x; i < (16 * CC) / 8; i += 256) d0[i] = s0[i];
  __syncthreads();
  f32x8 acc = {};
  for (int ks = 0; ks < 8; ++ks) {
    bf16x16 a = frag_a(lA + ks * 32, CC);
    acc = wmma_bf16(a, frag_b(cmt + (w * 16) * CC + ks * 32, CC), acc);
  }
  int j = w * 16 + (lane & 15);
  int mo = (lane < 16) ? 0 : 8;
#pragma unroll
  for (int v = 0; v < 8; ++v) {
    size_t o = (size_t)(n0 + v + mo) * DD + j;
    out[o] = out[o] + acc[v];
  }
}

extern "C" void kernel_launch(void* const* d_in, const int* in_sizes, int n_in,
                              void* d_out, int out_size, void* d_ws, size_t ws_size,
                              hipStream_t stream) {
  (void)n_in; (void)out_size; (void)ws_size;
  const float* node_memory  = (const float*)d_in[0];
  const float* last_update  = (const float*)d_in[1];
  const float* node_features= (const float*)d_in[2];
  const float* event_feat   = (const float*)d_in[3];
  const float* t            = (const float*)d_in[4];
  const int*   src          = (const int*)d_in[5];
  const int*   dst          = (const int*)d_in[6];
  const float* time_w       = (const float*)d_in[7];
  const float* time_b       = (const float*)d_in[8];
  const float* W_ih         = (const float*)d_in[9];
  const float* b_ih         = (const float*)d_in[10];
  const float* W_hh         = (const float*)d_in[11];
  const float* b_hh         = (const float*)d_in[12];
  const float* proj_W       = (const float*)d_in[13];
  const float* proj_b       = (const float*)d_in[14];
  const float* centroids    = (const float*)d_in[15];
  const int N = in_sizes[0] / DD;
  const int E = in_sizes[4];
  float* out = (float*)d_out;

  char* ws = (char*)d_ws;
  size_t off = 0;
  auto take = [&](size_t bytes) -> char* {
    char* p = ws + off; off = (off + bytes + 255) & ~(size_t)255; return p;
  };
  float* msg_sum  = (float*)take((size_t)N * KMSG * 4);
  float* cnt      = (float*)take((size_t)N * 4);
  bf16*  aggB     = (bf16*)take((size_t)N * KMSG * 2);
  bf16*  memB     = (bf16*)take((size_t)N * DD * 2);
  bf16*  WihB     = (bf16*)take((size_t)G3D * KMSG * 2);
  bf16*  WhhB     = (bf16*)take((size_t)G3D * DD * 2);
  bf16*  projWtB  = (bf16*)take((size_t)DD * DD * 2);
  bf16*  cenB     = (bf16*)take((size_t)CC * DD * 2);
  float* c_memory = (float*)take((size_t)CC * DD * 4);
  bf16*  cmt      = (bf16*)take((size_t)DD * CC * 2);
  // msg_sum region is dead after k_agg: alias simT (C*N f32) + ncB (N*C bf16) into it
  float* simT = msg_sum;
  bf16*  ncB  = (bf16*)((char*)msg_sum + (size_t)CC * N * 4);

  hipMemsetAsync(msg_sum, 0, (size_t)N * KMSG * 4, stream);
  hipMemsetAsync(cnt, 0, (size_t)N * 4, stream);

  // weight prep (bf16, B-friendly layouts)
  k_cvt<<<(G3D * KMSG + 255) / 256, 256, 0, stream>>>(W_ih, WihB, G3D * KMSG);
  k_cvt<<<(G3D * DD + 255) / 256, 256, 0, stream>>>(W_hh, WhhB, G3D * DD);
  k_cvt<<<((N * DD) + 255) / 256, 256, 0, stream>>>(node_memory, memB, N * DD);
  k_projt<<<(DD * DD + 255) / 256, 256, 0, stream>>>(proj_W, projWtB, DD * DD);
  k_cen<<<CC, DD, 0, stream>>>(centroids, cenB);

  // message scatter + mean aggregate
  k_scatter<<<E, 256, 0, stream>>>(node_memory, last_update, event_feat, t, src, dst,
                                   time_w, time_b, msg_sum, cnt, E);
  k_agg<<<((N * KMSG) + 255) / 256, 256, 0, stream>>>(msg_sum, cnt, aggB, N * KMSG);

  // GRU memory update -> d_out holds new_memory
  k_gru<<<N / 16, 256, 0, stream>>>(aggB, memB, WihB, WhhB, b_ih, b_hh,
                                    node_memory, cnt, out);

  // projection + cosine sim + row sparsemax (writes simT, ncB)
  k_comm<<<N / 16, 256, 0, stream>>>(out, node_features, projWtB, proj_b, cenB,
                                     simT, ncB, N);

  // column sparsemax fused with c_memory = cn_membership @ new_memory
  k_colsp<<<CC, 1024, 0, stream>>>(simT, out, c_memory, N);
  k_cmt<<<(DD * CC + 255) / 256, 256, 0, stream>>>(c_memory, cmt, DD * CC);

  // emb = new_memory + nc_membership @ c_memory (in-place add on d_out)
  k_emb<<<N / 16, 256, 0, stream>>>(ncB, cmt, out);
}
